// POPSSRingAttentionLayer_72112500900685
// MI455X (gfx1250) — matrix operand order repaired and checked by reading, not compiled, not run on hardware
//
#include <hip/hip_runtime.h>

typedef __attribute__((ext_vector_type(16))) __bf16 bf16x16;
typedef __attribute__((ext_vector_type(8)))  __bf16 bf16x8;
typedef __attribute__((ext_vector_type(8)))  float  f32x8;

namespace {
constexpr int Bn = 2, Sn = 4096, Hn = 16, Dn = 128;
constexpr int BS = 512;     // reference BLOCK_SIZE (per-block softmax granularity)
constexpr int MT = 128;     // query rows per workgroup
constexpr int NC = 64;      // keys per inner chunk (8 chunks per softmax block)
constexpr int QSTR = 136;   // padded LDS strides (bank-conflict avoidance)
constexpr int KSTR = 136;
constexpr int VSTR = 72;
constexpr int PSTR = 72;
}

__device__ __forceinline__ bf16x16 cat16(bf16x8 lo, bf16x8 hi) {
  return __builtin_shufflevector(lo, hi, 0,1,2,3,4,5,6,7,8,9,10,11,12,13,14,15);
}

__device__ __forceinline__ f32x8 wmma_bf16(bf16x16 a, bf16x16 b, f32x8 c) {
  // v_wmma_f32_16x16x32_bf16: (neg_a, A, neg_b, B, c_mod, C, reuse_a, reuse_b)
  return __builtin_amdgcn_wmma_f32_16x16x32_bf16(false, a, false, b, (short)0, c,
                                                 false, false);
}

__global__ void __launch_bounds__(256)
block_attn_wmma(const float* __restrict__ Qg, const float* __restrict__ Kg,
                const float* __restrict__ Vg, float* __restrict__ Og) {
  __shared__ __align__(16) __bf16 sQ[MT * QSTR];         // Q tile, row-major, pre-scaled
  __shared__ __align__(16) __bf16 sK[2][NC * KSTR];      // K chunk ping-pong (key x d)
  __shared__ __align__(16) __bf16 sV[2][Dn * VSTR];      // V chunk ping-pong, TRANSPOSED
  __shared__ __align__(16) __bf16 sP[8 * 16 * PSTR];     // per-wave P staging (C->A relayout)

  const int tid  = threadIdx.x;
  const int lane = tid & 31;
  const int wv   = tid >> 5;        // wave id 0..7 -> query rows wv*16..wv*16+15
  const int half = lane >> 4;       // 16-lane half (WMMA layout)
  const int l16  = lane & 15;

  const int qt   = blockIdx.x & 31;           // 32 query tiles of 128
  const int h    = (blockIdx.x >> 5) & 15;
  const int b    = blockIdx.x >> 9;
  const int qblk = qt >> 2;                   // 512-row block index
  const int q0   = qt * MT;
  const int nkb  = qblk + 1;                  // key blocks visible to this tile
  const int ntot = nkb * (BS / NC);           // total 64-key chunks

  const float scale = 0.08838834764831845f;   // 1/sqrt(128), folded into Q

  // ---- Load + convert Q tile (f32 -> bf16, scaled) ----
  {
    const int col = (tid & 31) * 4;
    const int r0  = tid >> 5;
    #pragma unroll
    for (int it = 0; it < 16; ++it) {
      const int r = r0 + it * 8;
      const float4 qv =
          *(const float4*)(Qg + (((size_t)b * Sn + (q0 + r)) * Hn + h) * Dn + col);
      union { __bf16 hh[4]; uint2 u; } pk;
      pk.hh[0] = (__bf16)(qv.x * scale);
      pk.hh[1] = (__bf16)(qv.y * scale);
      pk.hh[2] = (__bf16)(qv.z * scale);
      pk.hh[3] = (__bf16)(qv.w * scale);
      *(uint2*)&sQ[r * QSTR + col] = pk.u;     // one ds_store_b64 per float4
    }
  }

  // ---- K/V chunk filler: global f32 -> bf16 LDS (K row-major, V transposed) ----
  auto load_chunk = [&](int cc, int bufsel) {
    const int k0  = cc * NC;
    const int col = (tid & 31) * 4;
    const int r0  = tid >> 5;
    #pragma unroll
    for (int it = 0; it < 8; ++it) {
      const int r = r0 + it * 8;
      const size_t base = (((size_t)b * Sn + (k0 + r)) * Hn + h) * Dn + col;
      const float4 kv = *(const float4*)(Kg + base);
      union { __bf16 hh[4]; uint2 u; } pk;
      pk.hh[0] = (__bf16)kv.x; pk.hh[1] = (__bf16)kv.y;
      pk.hh[2] = (__bf16)kv.z; pk.hh[3] = (__bf16)kv.w;
      *(uint2*)&sK[bufsel][r * KSTR + col] = pk.u;   // one ds_store_b64
      const float4 vv = *(const float4*)(Vg + base);
      sV[bufsel][(col + 0) * VSTR + r] = (__bf16)vv.x;   // transposed scatter
      sV[bufsel][(col + 1) * VSTR + r] = (__bf16)vv.y;
      sV[bufsel][(col + 2) * VSTR + r] = (__bf16)vv.z;
      sV[bufsel][(col + 3) * VSTR + r] = (__bf16)vv.w;
    }
    // speculative prefetch of the chunk after next (global_prefetch_b8)
    const int pk0 = k0 + NC;
    if (pk0 < ntot * NC) {
      const size_t pb = (((size_t)b * Sn + (pk0 + r0)) * Hn + h) * Dn + col;
      __builtin_prefetch(Kg + pb, 0, 0);
      __builtin_prefetch(Vg + pb, 0, 0);
    }
  };

  // prime the pipeline with chunk 0
  load_chunk(0, 0);

  // ---- Hoist Q A-fragments for this wave's 16 rows (constant all kernel) ----
  __syncthreads();
  bf16x16 qf[4];
  {
    const int row = wv * 16 + l16;            // A layout: lane = M row
    #pragma unroll
    for (int ds = 0; ds < 4; ++ds) {
      const int dlo = ds * 32 + (half ? 8 : 0);
      bf16x8 lo = *(const bf16x8*)&sQ[row * QSTR + dlo];
      bf16x8 hi = *(const bf16x8*)&sQ[row * QSTR + dlo + 16];
      qf[ds] = cat16(lo, hi);
    }
  }

  f32x8 ofin[8];
  #pragma unroll
  for (int t = 0; t < 8; ++t) ofin[t] = f32x8{};

  f32x8 oacc[8];
  float mrow[8], lrow[8];

  for (int cc = 0; cc < ntot; ++cc) {
    const int j  = cc >> 3;                   // softmax block index
    const int k0 = cc * NC;
    if ((cc & 7) == 0) {                      // new key block: reset softmax state
      #pragma unroll
      for (int t = 0; t < 8; ++t) oacc[t] = f32x8{};
      #pragma unroll
      for (int r = 0; r < 8; ++r) { mrow[r] = -1e30f; lrow[r] = 0.0f; }
    }

    __syncthreads();     // chunk cc resident; buffer (cc+1)&1 free for refill
    if (cc + 1 < ntot) load_chunk(cc + 1, (cc + 1) & 1);  // overlaps compute below
    const int bs = cc & 1;

    // ---- Scores: 16 q-rows x 64 keys, contraction over D=128 ----
    f32x8 sc[4];
    #pragma unroll
    for (int nt = 0; nt < 4; ++nt) {
      f32x8 acc = f32x8{};
      const int krow = nt * 16 + l16;         // B layout: lane = N column (key)
      #pragma unroll
      for (int ds = 0; ds < 4; ++ds) {
        const int dlo = ds * 32 + half * 8;   // B: lanes0-15 K0-7,16-23, lanes16-31 K8-15,24-31
        bf16x8 lo = *(const bf16x8*)&sK[bs][krow * KSTR + dlo];
        bf16x8 hi = *(const bf16x8*)&sK[bs][krow * KSTR + dlo + 16];
        acc = wmma_bf16(qf[ds], cat16(lo, hi), acc);
      }
      sc[nt] = acc;
    }

    // ---- Causal mask inside the diagonal block (finite sentinel, no NaNs) ----
    if (j == qblk) {
      #pragma unroll
      for (int nt = 0; nt < 4; ++nt) {
        const int kg = k0 + nt * 16 + l16;
        #pragma unroll
        for (int r = 0; r < 8; ++r) {
          const int qg = q0 + wv * 16 + 8 * half + r;
          if (kg > qg) sc[nt][r] = -1e30f;
        }
      }
    }

    // ---- Online softmax update within this key block ----
    #pragma unroll
    for (int r = 0; r < 8; ++r) {
      float mx = fmaxf(fmaxf(sc[0][r], sc[1][r]), fmaxf(sc[2][r], sc[3][r]));
      #pragma unroll
      for (int off = 1; off < 16; off <<= 1)   // 16-lane row reduction per half
        mx = fmaxf(mx, __shfl_xor(mx, off, 32));
      const float mn    = fmaxf(mrow[r], mx);
      const float alpha = __expf(mrow[r] - mn);
      mrow[r] = mn;
      lrow[r] *= alpha;
      #pragma unroll
      for (int t = 0; t < 8; ++t) oacc[t][r] *= alpha;
      float rs = 0.0f;
      #pragma unroll
      for (int nt = 0; nt < 4; ++nt) {
        const float p = __expf(sc[nt][r] - mn);
        rs += p;
        sP[(wv * 16 + 8 * half + r) * PSTR + nt * 16 + l16] = (__bf16)p;
      }
      #pragma unroll
      for (int off = 1; off < 16; off <<= 1) rs += __shfl_xor(rs, off, 32);
      lrow[r] += rs;
    }

    // ---- P·V: contraction over the 64 keys (A from sP, B from transposed sV) ----
    #pragma unroll
    for (int f = 0; f < 2; ++f) {
      const int prow = wv * 16 + l16;
      const int klo  = f * 32 + (half ? 8 : 0);
      bf16x8 plo = *(const bf16x8*)&sP[prow * PSTR + klo];
      bf16x8 phi = *(const bf16x8*)&sP[prow * PSTR + klo + 16];
      const bf16x16 afrag = cat16(plo, phi);
      #pragma unroll
      for (int t = 0; t < 8; ++t) {
        const int drow = t * 16 + l16;
        const int kb   = f * 32 + half * 8;
        bf16x8 blo = *(const bf16x8*)&sV[bs][drow * VSTR + kb];
        bf16x8 bhi = *(const bf16x8*)&sV[bs][drow * VSTR + kb + 16];
        oacc[t] = wmma_bf16(afrag, cat16(blo, bhi), oacc[t]);
      }
    }

    // ---- End of key block: fold softmax-normalized output into running sum ----
    if ((cc & 7) == 7) {
      #pragma unroll
      for (int r = 0; r < 8; ++r) {
        const float inv = 1.0f / lrow[r];
        #pragma unroll
        for (int t = 0; t < 8; ++t) ofin[t][r] += oacc[t][r] * inv;
      }
    }
  }

  // ---- Average over blocks (reference's norm == i+1) and store ----
  const float avg = 1.0f / (float)nkb;
  #pragma unroll
  for (int t = 0; t < 8; ++t) {
    const int d = t * 16 + l16;
    #pragma unroll
    for (int r = 0; r < 8; ++r) {
      const int qg = q0 + wv * 16 + 8 * half + r;
      Og[(((size_t)b * Sn + qg) * Hn + h) * Dn + d] = ofin[t][r] * avg;
    }
  }
}

extern "C" void kernel_launch(void* const* d_in, const int* in_sizes, int n_in,
                              void* d_out, int out_size, void* d_ws, size_t ws_size,
                              hipStream_t stream) {
  const float* q = (const float*)d_in[0];
  const float* k = (const float*)d_in[1];
  const float* v = (const float*)d_in[2];
  float* o = (float*)d_out;
  const int grid = Bn * Hn * (Sn / MT);   // 1024 workgroups
  block_attn_wmma<<<grid, 256, 0, stream>>>(q, k, v, o);
}
